// MoELayerV2_56727928046139
// MI455X (gfx1250) — compile-verified, matrix operand order checked by reading, not run on hardware
//
#include <hip/hip_runtime.h>

#define N_TOK 4096
#define DM    2048
#define IN_   1024
#define NE    8

typedef __attribute__((ext_vector_type(16))) __bf16 v16bf;
typedef __attribute__((ext_vector_type(8)))  float  v8f;
typedef __attribute__((ext_vector_type(4)))  unsigned int u32x4;
typedef __attribute__((ext_vector_type(2)))  unsigned int u32x2;
typedef __attribute__((ext_vector_type(4)))  float  f32x4;

union Frag { v16bf v; u32x4 q[2]; };

__device__ __forceinline__ unsigned short f2bf(float f) {
  unsigned int u = __builtin_bit_cast(unsigned int, f);
  unsigned int r = u + 0x7FFFu + ((u >> 16) & 1u);   // round-to-nearest-even
  return (unsigned short)(r >> 16);
}

#if __has_builtin(__builtin_amdgcn_cvt_pk_bf16_f32)
__device__ __forceinline__ unsigned pack2bf(float a, float b) {
  typedef __attribute__((ext_vector_type(2))) __bf16 v2bf;
  v2bf r = __builtin_amdgcn_cvt_pk_bf16_f32(a, b);
  return __builtin_bit_cast(unsigned, r);
}
#else
__device__ __forceinline__ unsigned pack2bf(float a, float b) {
  return (unsigned)f2bf(a) | ((unsigned)f2bf(b) << 16);
}
#endif

// Async copy 16B from global (bf16 data) to LDS, tracked by ASYNCcnt.
// GVS mode: saddr = SGPR64 base, vaddr = per-lane 32-bit byte offset.
__device__ __forceinline__ void async_g2l_b128(unsigned lds_addr,
                                               unsigned gbyte_off,
                                               const void* sbase) {
  asm volatile("global_load_async_to_lds_b128 %0, %1, %2"
               :
               : "v"(lds_addr), "v"(gbyte_off), "s"(sbase)
               : "memory");
}
__device__ __forceinline__ void wait_async0() {
  asm volatile("s_wait_asynccnt 0x0" ::: "memory");
}

// ---------------- init: zero output, counters, loss sums; identity list ----
__global__ __launch_bounds__(256)
void init_kernel(float* __restrict__ out, int* __restrict__ ident,
                 int* __restrict__ counts, float* __restrict__ sums) {
  size_t i = (size_t)blockIdx.x * 256 + threadIdx.x;
  if (i < (size_t)N_TOK * DM + 1) out[i] = 0.0f;
  if (i < N_TOK) ident[i] = (int)i;
  if (i < 9)  counts[i] = (i == 8) ? N_TOK : 0;   // counts[8] = shared-expert count
  if (i < 17) sums[i] = 0.0f;                     // [0..7]=mask, [8..15]=score, [16]=z^2
}

// ---------------- x fp32 -> bf16 ------------------------------------------
__global__ __launch_bounds__(256)
void cvt_kernel(const float* __restrict__ x, unsigned short* __restrict__ xb) {
  size_t i = ((size_t)blockIdx.x * 256 + threadIdx.x) * 4;
  f32x4 v = *(const f32x4*)(x + i);
  u32x2 o;
  o[0] = pack2bf(v[0], v[1]);
  o[1] = pack2bf(v[2], v[3]);
  *(u32x2*)(xb + i) = o;
}

// ---------------- router: one wave per token ------------------------------
__global__ __launch_bounds__(256)
void router_kernel(const float* __restrict__ x, const float* __restrict__ gw,
                   float* __restrict__ combine, int* __restrict__ counts,
                   int* __restrict__ etok, float* __restrict__ sums) {
  int wid = threadIdx.x >> 5, lane = threadIdx.x & 31;
  int tok = blockIdx.x * 8 + wid;
  const float* xr = x + (size_t)tok * DM;
  float lg[NE];
  for (int e = 0; e < NE; e++) {
    const float* g = gw + (size_t)e * DM;
    float p = 0.f;
    for (int k = lane; k < DM; k += 32) p += xr[k] * g[k];
    for (int off = 16; off > 0; off >>= 1) p += __shfl_xor(p, off, 32);
    lg[e] = p;
  }
  if (lane == 0) {
    float mx = lg[0];
    for (int e = 1; e < NE; e++) mx = fmaxf(mx, lg[e]);
    float sc[NE]; float se = 0.f;
    for (int e = 0; e < NE; e++) { sc[e] = __expf(lg[e] - mx); se += sc[e]; }
    float z = mx + __logf(se);
    float inv_se = 1.f / se;
    for (int e = 0; e < NE; e++) sc[e] *= inv_se;
    int i1 = 0;
    for (int e = 1; e < NE; e++) if (sc[e] > sc[i1]) i1 = e;
    int i2 = (i1 == 0) ? 1 : 0;
    for (int e = 0; e < NE; e++) if (e != i1 && sc[e] > sc[i2]) i2 = e;
    float s1 = sc[i1], s2 = sc[i2], inv = 1.f / (s1 + s2);
    for (int e = 0; e < NE; e++) combine[(size_t)tok * NE + e] = 0.f;
    combine[(size_t)tok * NE + i1] = s1 * inv;
    combine[(size_t)tok * NE + i2] = s2 * inv;
    for (int e = 0; e < NE; e++) atomicAdd(&sums[NE + e], sc[e]);
    atomicAdd(&sums[i1], 1.f);
    atomicAdd(&sums[i2], 1.f);
    atomicAdd(&sums[16], z * z);
    int p1 = atomicAdd(&counts[i1], 1); etok[(size_t)i1 * N_TOK + p1] = tok;
    int p2 = atomicAdd(&counts[i2], 1); etok[(size_t)i2 * N_TOK + p2] = tok;
  }
}

// ---------------- loss scalar ---------------------------------------------
__global__ void loss_kernel(const float* __restrict__ sums, float* __restrict__ out) {
  if (threadIdx.x == 0 && blockIdx.x == 0) {
    const float invN = 1.0f / (float)N_TOK;
    float aux = 0.f;
    for (int e = 0; e < NE; e++) aux += (sums[e] * invN) * (sums[NE + e] * invN);
    out[(size_t)N_TOK * DM] = 0.01f * (float)NE * aux + 0.001f * (sums[16] * invN);
  }
}

// ---------------- up projection: h = silu(x@W1^T) * (x@W3^T) --------------
// Tile: 128 gathered tokens x 64 inter cols, K-chunk 32 over D.
// Double-buffered LDS; A tile via async global->LDS, B tiles via VGPR+cvt.
__global__ __launch_bounds__(256)
void moe_up_kernel(const unsigned short* __restrict__ xb,
                   const int* __restrict__ tokensBase,
                   const int* __restrict__ countBase,
                   const float* __restrict__ w1Base,
                   const float* __restrict__ w3Base,
                   unsigned short* __restrict__ hbuf,
                   int tokStride, unsigned long long wStride, int hRowBase) {
  const unsigned A_HALF = 128 * 40, B_HALF = 64 * 40;
  __shared__ __align__(16) unsigned short lA[2 * 128 * 40];
  __shared__ __align__(16) unsigned short lB1[2 * 64 * 40];
  __shared__ __align__(16) unsigned short lB3[2 * 64 * 40];

  const int e = blockIdx.z;
  const int count = countBase[e];
  const int tileM = blockIdx.y;
  if (tileM * 128 >= count) return;
  const int nBase = blockIdx.x * 64;
  const int* tokens = tokensBase + (size_t)e * tokStride;
  const float* W1 = w1Base + (size_t)e * wStride;
  const float* W3 = w3Base + (size_t)e * wStride;
  const int hRow = hRowBase + e * N_TOK;

  const int tid = threadIdx.x;
  const int wid = tid >> 5, lane = tid & 31;
  const int wm = wid & 3, wn = wid >> 2;
  const int lr = lane & 15, hs = lane >> 4;

  // A-tile staging: 128 rows x 64B = 512 chunks of 16B; 2 per thread.
  const int r0 = tid >> 2, part = tid & 3;
  const int slot0 = tileM * 128 + r0;
  const int slot1 = slot0 + 64;
  const int tokA = tokens[slot0 < count ? slot0 : 0];
  const int tokB = tokens[slot1 < count ? slot1 : 0];
  const unsigned ldsA0 = (unsigned)(size_t)(lA + r0 * 40 + part * 8);
  const unsigned ldsA1 = (unsigned)(size_t)(lA + (r0 + 64) * 40 + part * 8);
  const unsigned A_BYTES = A_HALF * 2;
  const unsigned gA0 = ((unsigned)tokA * DM + part * 8) * 2u;
  const unsigned gA1 = ((unsigned)tokB * DM + part * 8) * 2u;

  // B-tile staging: thread -> (n0, kk0) and (n0+32, kk0)
  const int n0 = tid >> 3;             // 0..31
  const int kk0 = (tid & 7) << 2;      // 0..28
  const size_t goBase = (size_t)(nBase + n0) * DM + kk0;
  const size_t goStep = (size_t)32 * DM;

  v8f accg[2][2], accu[2][2];
  for (int a = 0; a < 2; a++)
    for (int b = 0; b < 2; b++)
      for (int j = 0; j < 8; j++) { accg[a][b][j] = 0.f; accu[a][b][j] = 0.f; }

  // ---- prologue: stage K-chunk 0 into buffer 0 ----
  async_g2l_b128(ldsA0, gA0, xb);
  async_g2l_b128(ldsA1, gA1, xb);
  {
    f32x4 f1a = *(const f32x4*)(W1 + goBase);
    f32x4 f3a = *(const f32x4*)(W3 + goBase);
    f32x4 f1b = *(const f32x4*)(W1 + goBase + goStep);
    f32x4 f3b = *(const f32x4*)(W3 + goBase + goStep);
    u32x2 p;
    p[0] = pack2bf(f1a[0], f1a[1]); p[1] = pack2bf(f1a[2], f1a[3]);
    *(u32x2*)(lB1 + n0 * 40 + kk0) = p;
    p[0] = pack2bf(f3a[0], f3a[1]); p[1] = pack2bf(f3a[2], f3a[3]);
    *(u32x2*)(lB3 + n0 * 40 + kk0) = p;
    p[0] = pack2bf(f1b[0], f1b[1]); p[1] = pack2bf(f1b[2], f1b[3]);
    *(u32x2*)(lB1 + (n0 + 32) * 40 + kk0) = p;
    p[0] = pack2bf(f3b[0], f3b[1]); p[1] = pack2bf(f3b[2], f3b[3]);
    *(u32x2*)(lB3 + (n0 + 32) * 40 + kk0) = p;
  }

  const int T = DM / 32;
  for (int t = 0; t < T; t++) {
    wait_async0();
    __syncthreads();
    const int cur = t & 1, nxt = cur ^ 1;
    const bool more = (t + 1 < T);
    const int k1 = (t + 1) * 32;

    // fire async A loads for next buffer
    if (more) {
      async_g2l_b128(ldsA0 + (unsigned)nxt * A_BYTES, gA0 + (unsigned)k1 * 2u, xb);
      async_g2l_b128(ldsA1 + (unsigned)nxt * A_BYTES, gA1 + (unsigned)k1 * 2u, xb);
    }
    // issue next weight loads into registers (overlaps WMMA below)
    f32x4 f1a, f3a, f1b, f3b;
    if (more) {
      const size_t go = goBase + k1;
      f1a = *(const f32x4*)(W1 + go);
      f3a = *(const f32x4*)(W3 + go);
      f1b = *(const f32x4*)(W1 + go + goStep);
      f3b = *(const f32x4*)(W3 + go + goStep);
      if (t + 2 < T) {
        __builtin_prefetch(W1 + go + 32);
        __builtin_prefetch(W3 + go + 32);
      }
    }

    // fragments from current buffer (ds_loads precede any ds_store)
    const unsigned short* Ab  = lA  + cur * A_HALF;
    const unsigned short* B1b = lB1 + cur * B_HALF;
    const unsigned short* B3b = lB3 + cur * B_HALF;
    Frag a[2], b1[2], b3[2];
    for (int mi = 0; mi < 2; mi++) {
      const unsigned short* p = Ab + (wm * 32 + mi * 16 + lr) * 40 + hs * 8;
      a[mi].q[0] = *(const u32x4*)p;
      a[mi].q[1] = *(const u32x4*)(p + 16);
    }
    for (int ni = 0; ni < 2; ni++) {
      const unsigned short* p1 = B1b + (wn * 32 + ni * 16 + lr) * 40 + hs * 8;
      b1[ni].q[0] = *(const u32x4*)p1;
      b1[ni].q[1] = *(const u32x4*)(p1 + 16);
      const unsigned short* p3 = B3b + (wn * 32 + ni * 16 + lr) * 40 + hs * 8;
      b3[ni].q[0] = *(const u32x4*)p3;
      b3[ni].q[1] = *(const u32x4*)(p3 + 16);
    }
    for (int mi = 0; mi < 2; mi++)
      for (int ni = 0; ni < 2; ni++) {
        accg[mi][ni] = __builtin_amdgcn_wmma_f32_16x16x32_bf16(
            false, a[mi].v, false, b1[ni].v, (short)0, accg[mi][ni], false, false);
        accu[mi][ni] = __builtin_amdgcn_wmma_f32_16x16x32_bf16(
            false, a[mi].v, false, b3[ni].v, (short)0, accu[mi][ni], false, false);
      }

    // convert + store staged weights into next buffer (after frag loads)
    if (more) {
      unsigned short* D1 = lB1 + nxt * B_HALF;
      unsigned short* D3 = lB3 + nxt * B_HALF;
      u32x2 p;
      p[0] = pack2bf(f1a[0], f1a[1]); p[1] = pack2bf(f1a[2], f1a[3]);
      *(u32x2*)(D1 + n0 * 40 + kk0) = p;
      p[0] = pack2bf(f3a[0], f3a[1]); p[1] = pack2bf(f3a[2], f3a[3]);
      *(u32x2*)(D3 + n0 * 40 + kk0) = p;
      p[0] = pack2bf(f1b[0], f1b[1]); p[1] = pack2bf(f1b[2], f1b[3]);
      *(u32x2*)(D1 + (n0 + 32) * 40 + kk0) = p;
      p[0] = pack2bf(f3b[0], f3b[1]); p[1] = pack2bf(f3b[2], f3b[3]);
      *(u32x2*)(D3 + (n0 + 32) * 40 + kk0) = p;
    }
  }

  for (int mi = 0; mi < 2; mi++)
    for (int ni = 0; ni < 2; ni++) {
      int col = nBase + wn * 32 + ni * 16 + lr;
      int base = tileM * 128 + wm * 32 + mi * 16 + hs * 8;
      for (int j = 0; j < 8; j++) {
        int slot = base + j;
        if (slot < count) {
          float g = accg[mi][ni][j];
          float u = accu[mi][ni][j];
          float s = g * (1.0f / (1.0f + __expf(-g)));
          hbuf[(size_t)(hRow + slot) * IN_ + col] = f2bf(s * u);
        }
      }
    }
}

// ---------------- down projection: out += combine * (h @ W2^T) ------------
__global__ __launch_bounds__(256)
void moe_down_kernel(const unsigned short* __restrict__ hbuf,
                     const int* __restrict__ tokensBase,
                     const int* __restrict__ countBase,
                     const float* __restrict__ w2Base,
                     const float* __restrict__ combine,
                     float* __restrict__ out,
                     int tokStride, unsigned long long wStride,
                     int hRowBase, int useCombine) {
  const unsigned A_HALF = 128 * 40, B_HALF = 64 * 40;
  __shared__ __align__(16) unsigned short lA[2 * 128 * 40];
  __shared__ __align__(16) unsigned short lB[2 * 64 * 40];

  const int e = blockIdx.z;
  const int count = countBase[e];
  const int tileM = blockIdx.y;
  if (tileM * 128 >= count) return;
  const int nBase = blockIdx.x * 64;
  const int* tokens = tokensBase + (size_t)e * tokStride;
  const float* W2 = w2Base + (size_t)e * wStride;
  const int hRow = hRowBase + e * N_TOK;

  const int tid = threadIdx.x;
  const int wid = tid >> 5, lane = tid & 31;
  const int wm = wid & 3, wn = wid >> 2;
  const int lr = lane & 15, hs = lane >> 4;

  const int r0 = tid >> 2, part = tid & 3;
  const unsigned ldsA0 = (unsigned)(size_t)(lA + r0 * 40 + part * 8);
  const unsigned ldsA1 = (unsigned)(size_t)(lA + (r0 + 64) * 40 + part * 8);
  const unsigned A_BYTES = A_HALF * 2;
  const unsigned rowBase = (unsigned)(hRow + tileM * 128);
  const unsigned gA0 = ((rowBase + r0) * IN_ + part * 8) * 2u;
  const unsigned gA1 = ((rowBase + r0 + 64) * IN_ + part * 8) * 2u;

  const int n0 = tid >> 3;
  const int kk0 = (tid & 7) << 2;
  const size_t goBase = (size_t)(nBase + n0) * IN_ + kk0;
  const size_t goStep = (size_t)32 * IN_;

  v8f acc[2][2];
  for (int a = 0; a < 2; a++)
    for (int b = 0; b < 2; b++)
      for (int j = 0; j < 8; j++) acc[a][b][j] = 0.f;

  // prologue
  async_g2l_b128(ldsA0, gA0, hbuf);
  async_g2l_b128(ldsA1, gA1, hbuf);
  {
    f32x4 fa = *(const f32x4*)(W2 + goBase);
    f32x4 fb = *(const f32x4*)(W2 + goBase + goStep);
    u32x2 p;
    p[0] = pack2bf(fa[0], fa[1]); p[1] = pack2bf(fa[2], fa[3]);
    *(u32x2*)(lB + n0 * 40 + kk0) = p;
    p[0] = pack2bf(fb[0], fb[1]); p[1] = pack2bf(fb[2], fb[3]);
    *(u32x2*)(lB + (n0 + 32) * 40 + kk0) = p;
  }

  const int T = IN_ / 32;
  for (int t = 0; t < T; t++) {
    wait_async0();
    __syncthreads();
    const int cur = t & 1, nxt = cur ^ 1;
    const bool more = (t + 1 < T);
    const int k1 = (t + 1) * 32;

    if (more) {
      async_g2l_b128(ldsA0 + (unsigned)nxt * A_BYTES, gA0 + (unsigned)k1 * 2u, hbuf);
      async_g2l_b128(ldsA1 + (unsigned)nxt * A_BYTES, gA1 + (unsigned)k1 * 2u, hbuf);
    }
    f32x4 fa, fb;
    if (more) {
      const size_t go = goBase + k1;
      fa = *(const f32x4*)(W2 + go);
      fb = *(const f32x4*)(W2 + go + goStep);
      if (t + 2 < T) __builtin_prefetch(W2 + go + 32);
    }

    const unsigned short* Ab = lA + cur * A_HALF;
    const unsigned short* Bb = lB + cur * B_HALF;
    Frag a[2], b[2];
    for (int mi = 0; mi < 2; mi++) {
      const unsigned short* p = Ab + (wm * 32 + mi * 16 + lr) * 40 + hs * 8;
      a[mi].q[0] = *(const u32x4*)p;
      a[mi].q[1] = *(const u32x4*)(p + 16);
    }
    for (int ni = 0; ni < 2; ni++) {
      const unsigned short* p = Bb + (wn * 32 + ni * 16 + lr) * 40 + hs * 8;
      b[ni].q[0] = *(const u32x4*)p;
      b[ni].q[1] = *(const u32x4*)(p + 16);
    }
    for (int mi = 0; mi < 2; mi++)
      for (int ni = 0; ni < 2; ni++)
        acc[mi][ni] = __builtin_amdgcn_wmma_f32_16x16x32_bf16(
            false, a[mi].v, false, b[ni].v, (short)0, acc[mi][ni], false, false);

    if (more) {
      unsigned short* D = lB + nxt * B_HALF;
      u32x2 p;
      p[0] = pack2bf(fa[0], fa[1]); p[1] = pack2bf(fa[2], fa[3]);
      *(u32x2*)(D + n0 * 40 + kk0) = p;
      p[0] = pack2bf(fb[0], fb[1]); p[1] = pack2bf(fb[2], fb[3]);
      *(u32x2*)(D + (n0 + 32) * 40 + kk0) = p;
    }
  }

  for (int mi = 0; mi < 2; mi++)
    for (int ni = 0; ni < 2; ni++) {
      int col = nBase + wn * 32 + ni * 16 + lr;
      int base = tileM * 128 + wm * 32 + mi * 16 + hs * 8;
      for (int j = 0; j < 8; j++) {
        int slot = base + j;
        if (slot < count) {
          int tok = tokens[slot];
          if (useCombine) {
            float w = combine[(size_t)tok * NE + e];
            atomicAdd(&out[(size_t)tok * DM + col], w * acc[mi][ni][j]);
          } else {
            // shared expert: stream-ordered after expert pass, single writer
            size_t idx = (size_t)tok * DM + col;
            out[idx] = out[idx] + acc[mi][ni][j];
          }
        }
      }
    }
}

// ---------------- host-side launch ----------------------------------------
extern "C" void kernel_launch(void* const* d_in, const int* in_sizes, int n_in,
                              void* d_out, int out_size, void* d_ws, size_t ws_size,
                              hipStream_t stream) {
  (void)in_sizes; (void)n_in; (void)out_size; (void)ws_size;
  const float* x      = (const float*)d_in[0];
  const float* gate_w = (const float*)d_in[1];
  const float* w1     = (const float*)d_in[2];
  const float* w3     = (const float*)d_in[3];
  const float* w2     = (const float*)d_in[4];
  const float* sw1    = (const float*)d_in[5];
  const float* sw3    = (const float*)d_in[6];
  const float* sw2    = (const float*)d_in[7];
  float* out = (float*)d_out;

  char* ws = (char*)d_ws;
  size_t off = 0;
  unsigned short* xb   = (unsigned short*)(ws + off); off += (size_t)N_TOK * DM * 2;
  unsigned short* hbuf = (unsigned short*)(ws + off); off += (size_t)9 * N_TOK * IN_ * 2;
  float* combine = (float*)(ws + off); off += (size_t)N_TOK * NE * 4;
  int*   etok    = (int*)(ws + off);   off += (size_t)NE * N_TOK * 4;
  int*   ident   = (int*)(ws + off);   off += (size_t)N_TOK * 4;
  int*   counts  = (int*)(ws + off);   off += 64;
  float* sums    = (float*)(ws + off);

  init_kernel<<<32769, 256, 0, stream>>>(out, ident, counts, sums);
  cvt_kernel<<<8192, 256, 0, stream>>>(x, xb);
  router_kernel<<<512, 256, 0, stream>>>(x, gate_w, combine, counts, etok, sums);
  loss_kernel<<<1, 64, 0, stream>>>(sums, out);

  // routed experts: grid (inter tiles, token tiles, experts)
  moe_up_kernel<<<dim3(IN_ / 64, 32, NE), 256, 0, stream>>>(
      xb, etok, counts, w1, w3, hbuf,
      /*tokStride=*/N_TOK, /*wStride=*/(unsigned long long)IN_ * DM, /*hRowBase=*/0);
  // shared expert (all tokens, weight 1)
  moe_up_kernel<<<dim3(IN_ / 64, 32, 1), 256, 0, stream>>>(
      xb, ident, counts + 8, sw1, sw3, hbuf,
      /*tokStride=*/0, /*wStride=*/0ULL, /*hRowBase=*/NE * N_TOK);

  moe_down_kernel<<<dim3(DM / 64, 32, NE), 256, 0, stream>>>(
      hbuf, etok, counts, w2, combine, out,
      /*tokStride=*/N_TOK, /*wStride=*/(unsigned long long)DM * IN_,
      /*hRowBase=*/0, /*useCombine=*/1);
  moe_down_kernel<<<dim3(DM / 64, 32, 1), 256, 0, stream>>>(
      hbuf, ident, counts + 8, sw2, combine, out,
      /*tokStride=*/0, /*wStride=*/0ULL,
      /*hRowBase=*/NE * N_TOK, /*useCombine=*/0);
}
